// DynamicAUGRU_38276748542531
// MI455X (gfx1250) — compile-verified, matrix operand order checked by reading, not run on hardware
//
#include <hip/hip_runtime.h>
#include <hip/hip_bf16.h>

// ---- CDNA5 WMMA vector types ----
typedef __bf16 v8bf  __attribute__((ext_vector_type(8)));
typedef __bf16 v16bf __attribute__((ext_vector_type(16)));
typedef float  v8f   __attribute__((ext_vector_type(8)));

#define BATCH 4096
#define TT    200
#define II    128
#define HHD   128
#define G3    384
#define BM    32          // batch rows per block (2 M-tiles)
#define KS    264         // padded LDS row stride (bf16) -> 528B, breaks bank conflicts
#define NTHREADS 256      // 8 wave32 waves; wave w owns hidden cols 16w..16w+15

// Native CDNA5 v_tanh_f32 if the builtin exists; exp-based fallback otherwise.
__device__ __forceinline__ float tanh_hw(float x) {
#if __has_builtin(__builtin_amdgcn_tanhf)
    return __builtin_amdgcn_tanhf(x);
#else
    x = fminf(fmaxf(x, -12.0f), 12.0f);
    float e = __expf(2.0f * x);
    return (e - 1.0f) / (e + 1.0f);
#endif
}
__device__ __forceinline__ float sig_hw(float x) {
    return 0.5f + 0.5f * tanh_hw(0.5f * x);   // 1 trans op instead of exp+rcp
}

__device__ __forceinline__ v16bf ld_frag(const __bf16* p) {
    v8bf lo = *(const v8bf*)p;
    v8bf hi = *(const v8bf*)(p + 16);
    return __builtin_shufflevector(lo, hi, 0,1,2,3,4,5,6,7,8,9,10,11,12,13,14,15);
}
#define WMMA_BF16(A, Bm, C) \
    __builtin_amdgcn_wmma_f32_16x16x32_bf16(false, (A), false, (Bm), (short)0, (C), false, false)

__global__ __launch_bounds__(NTHREADS, 1)
void augru_fused_kernel(const float* __restrict__ inputs,   // (B, T, I)
                        const float* __restrict__ att,      // (B, T)
                        const int*   __restrict__ mask,     // (B, T)
                        const float* __restrict__ Wx,       // (3H, I)
                        const float* __restrict__ bx,       // (3H,)
                        const float* __restrict__ Wh,       // (3H, H)
                        const float* __restrict__ bh,       // (3H,)
                        float*       __restrict__ out)      // (B, H)
{
    __shared__ __bf16 W_lds[G3 * KS];        // 202752 B : rows g = [Wx[g] | Wh[g]] bf16
    __shared__ __bf16 A_lds[BM * KS];        //  16896 B : rows b = [x_t | h] bf16
    __shared__ float  att_s[2][BM];          // double-buffered per-row attention
    __shared__ int    m_s[2][BM];            // double-buffered per-row mask

    const int tid  = threadIdx.x;
    const int lane = tid & 31;
    const int wave = tid >> 5;
    const int ln   = lane & 15;    // N / row-in-16 index
    const int lh   = lane >> 4;    // half-wave select
    const int jcol = wave * 16;    // this wave's hidden-column range
    const int b0   = blockIdx.x * BM;

    // ---------------- prefetch x(0), att(0), mask(0) into registers ----------------
    float4 xr[4];
    #pragma unroll
    for (int k = 0; k < 4; ++k) {
        int f  = tid + (k << 8);          // 1024 float4 total, stride NTHREADS
        int r  = f >> 5;                  // 32 float4 per 128-wide row
        int c4 = (f & 31) << 2;
        xr[k] = *(const float4*)&inputs[((long)(b0 + r) * TT + 0) * II + c4];
    }
    float a_pf = 0.0f; int m_pf = 0;
    if (tid < BM) {
        a_pf = att[(long)(b0 + tid) * TT + 0];
        m_pf = mask[(long)(b0 + tid) * TT + 0];
    }

    // ---------------- weights -> LDS bf16 (overlaps with x(0) latency) -------------
    for (int idx = tid; idx < G3 * II / 4; idx += NTHREADS) {
        int g  = idx >> 5;
        int i4 = (idx & 31) * 4;
        float4 wx = ((const float4*)Wx)[idx];
        float4 wh = ((const float4*)Wh)[idx];
        __bf16* px = &W_lds[g * KS + i4];
        px[0] = (__bf16)wx.x; px[1] = (__bf16)wx.y; px[2] = (__bf16)wx.z; px[3] = (__bf16)wx.w;
        __bf16* ph = &W_lds[g * KS + 128 + i4];
        ph[0] = (__bf16)wh.x; ph[1] = (__bf16)wh.y; ph[2] = (__bf16)wh.z; ph[3] = (__bf16)wh.w;
    }
    // per-lane fused biases for this wave's columns (registers, constant over time)
    const int c = jcol + ln;
    const float bias_u  = bx[c]       + bh[c];
    const float bias_r  = bx[128 + c] + bh[128 + c];
    const float bias_x  = bx[256 + c];
    const float bias_hh = bh[256 + c];

    // h0 = 0 : zero the bf16 h-half of A
    for (int e = tid; e < BM * HHD; e += NTHREADS) {
        int r = e >> 7, j = e & 127;
        A_lds[r * KS + 128 + j] = (__bf16)0.0f;
    }
    __syncthreads();   // W_lds fully written before hoisting B fragments

    // ---------------- hoist ALL weight (B) fragments into registers ----------------
    // 3 gate-sets x 8 k-tiles x 8 VGPRs = 192 VGPRs; constant across all 200 steps.
    const __bf16* Bu  = &W_lds[(jcol + ln) * KS];
    const __bf16* Br  = &W_lds[(128 + jcol + ln) * KS];
    const __bf16* Bx2 = &W_lds[(256 + jcol + ln) * KS];
    v16bf Bfu[8], Bfr[8], Bfz[8];
    #pragma unroll
    for (int kt = 0; kt < 8; ++kt) {
        const int k0 = kt * 32 + lh * 8;
        Bfu[kt] = ld_frag(Bu  + k0);
        Bfr[kt] = ld_frag(Br  + k0);
        Bfz[kt] = ld_frag(Bx2 + k0);
    }

    float hreg[16];                    // persistent f32 hidden state (row mt*16+lh*8+v, col jcol+ln)
    #pragma unroll
    for (int i = 0; i < 16; ++i) hreg[i] = 0.0f;

    const __bf16* Abase0 = &A_lds[(0  + ln) * KS];
    const __bf16* Abase1 = &A_lds[(16 + ln) * KS];

    // ---------------- time loop ----------------
    for (int t = 0; t < TT; ++t) {
        const int buf = t & 1;
        // (a) convert prefetched x(t) -> A_lds x-half; stage att/mask
        #pragma unroll
        for (int k = 0; k < 4; ++k) {
            int f  = tid + (k << 8);
            int r  = f >> 5;
            int c4 = (f & 31) << 2;
            __bf16* p = &A_lds[r * KS + c4];
            p[0] = (__bf16)xr[k].x; p[1] = (__bf16)xr[k].y;
            p[2] = (__bf16)xr[k].z; p[3] = (__bf16)xr[k].w;
        }
        if (tid < BM) { att_s[buf][tid] = a_pf; m_s[buf][tid] = m_pf; }
        __syncthreads();

        // (b) prefetch x(t+1)/att/mask during GEMM (latency hidden behind WMMA)
        if (t + 1 < TT) {
            const int tn = t + 1;
            #pragma unroll
            for (int k = 0; k < 4; ++k) {
                int f  = tid + (k << 8);
                int r  = f >> 5;
                int c4 = (f & 31) << 2;
                xr[k] = *(const float4*)&inputs[((long)(b0 + r) * TT + tn) * II + c4];
            }
            if (tid < BM) {
                a_pf = att[(long)(b0 + tid) * TT + tn];
                m_pf = mask[(long)(b0 + tid) * TT + tn];
            }
        }

        // (c) GEMM: only A fragments come from LDS now (32 b128 loads / step);
        //     all B operands are register-resident.
        v8f au0 = {}, au1 = {}, ar0 = {}, ar1 = {};
        v8f ax0 = {}, ax1 = {}, ah0 = {}, ah1 = {};
        #pragma unroll
        for (int kt = 0; kt < 8; ++kt) {
            const int k0 = kt * 32 + lh * 8;
            v16bf a0 = ld_frag(Abase0 + k0);
            v16bf a1 = ld_frag(Abase1 + k0);
            au0 = WMMA_BF16(a0, Bfu[kt], au0);
            au1 = WMMA_BF16(a1, Bfu[kt], au1);
            ar0 = WMMA_BF16(a0, Bfr[kt], ar0);
            ar1 = WMMA_BF16(a1, Bfr[kt], ar1);
            if (kt < 4) {               // K 0..127  = x side  -> x_h gate
                ax0 = WMMA_BF16(a0, Bfz[kt], ax0);
                ax1 = WMMA_BF16(a1, Bfz[kt], ax1);
            } else {                    // K 128..255 = h side -> h_h gate
                ah0 = WMMA_BF16(a0, Bfz[kt], ah0);
                ah1 = WMMA_BF16(a1, Bfz[kt], ah1);
            }
        }
        __syncthreads();   // all waves done reading A before h write-back

        // (d) elementwise AUGRU update entirely in registers (3 trans ops/elem)
        #pragma unroll
        for (int mt = 0; mt < 2; ++mt) {
            #pragma unroll
            for (int v = 0; v < 8; ++v) {
                const int row = mt * 16 + lh * 8 + v;
                const float up = (mt ? au1[v] : au0[v]) + bias_u;
                const float rp = (mt ? ar1[v] : ar0[v]) + bias_r;
                const float xh = (mt ? ax1[v] : ax0[v]) + bias_x;
                const float hh = (mt ? ah1[v] : ah0[v]) + bias_hh;
                const float a  = att_s[buf][row];
                const int   m  = m_s[buf][row];
                float h  = hreg[mt * 8 + v];
                float u  = sig_hw(up);
                float rr = sig_hw(rp);
                float ht = tanh_hw(xh + rr * hh);
                float ua = a * u;
                float hn = (1.0f - ua) * h + ua * ht;
                h = m ? hn : h;
                hreg[mt * 8 + v] = h;
                A_lds[row * KS + 128 + jcol + ln] = (__bf16)h;  // bf16 copy for next GEMM
            }
        }
        // no barrier needed here: next iteration's barrier (after x conversion)
        // orders these h stores before any GEMM read.
    }

    // ---------------- final hidden state -> out (B, H) f32 ----------------
    #pragma unroll
    for (int mt = 0; mt < 2; ++mt) {
        #pragma unroll
        for (int v = 0; v < 8; ++v) {
            const int row = mt * 16 + lh * 8 + v;
            out[(long)(b0 + row) * HHD + jcol + ln] = hreg[mt * 8 + v];
        }
    }
}

extern "C" void kernel_launch(void* const* d_in, const int* in_sizes, int n_in,
                              void* d_out, int out_size, void* d_ws, size_t ws_size,
                              hipStream_t stream) {
    (void)in_sizes; (void)n_in; (void)d_ws; (void)ws_size; (void)out_size;
    const float* inputs = (const float*)d_in[0];
    const float* att    = (const float*)d_in[1];
    const int*   maskp  = (const int*)  d_in[2];
    const float* Wx     = (const float*)d_in[3];
    const float* bx     = (const float*)d_in[4];
    const float* Wh     = (const float*)d_in[5];
    const float* bh     = (const float*)d_in[6];
    float* out = (float*)d_out;

    dim3 grid(BATCH / BM);     // 128 independent batch-slice blocks
    dim3 block(NTHREADS);      // 8 wave32 waves
    augru_fused_kernel<<<grid, block, 0, stream>>>(inputs, att, maskp, Wx, bx, Wh, bh, out);
}
